// VQUNetBase_58428735095051
// MI455X (gfx1250) — matrix operand order repaired and checked by reading, not compiled, not run on hardware
//
#include <hip/hip_runtime.h>
#include <stdint.h>

// ---------------------------------------------------------------------------
// CDNA5 (gfx1250) wave32 WMMA types / helpers
// ---------------------------------------------------------------------------
typedef __bf16 bf16_t;
typedef bf16_t bf16x16 __attribute__((ext_vector_type(16)));
typedef float  f32x8   __attribute__((ext_vector_type(8)));
typedef int    v4i     __attribute__((vector_size(16)));

struct U32x8 { uint4 lo, hi; };

__device__ __forceinline__ bf16x16 as_bfrag(const U32x8& u) {
  return __builtin_bit_cast(bf16x16, u);
}

__device__ __forceinline__ unsigned short f32_bf16(float f) {
  unsigned int u = __builtin_bit_cast(unsigned int, f);
  u += 0x7FFFu + ((u >> 16) & 1u);          // round-to-nearest-even
  return (unsigned short)(u >> 16);
}

// ---------------------------------------------------------------------------
// Global -> LDS staging. Prefer the CDNA5 async-to-LDS path (ASYNCcnt) when
// the builtins exist; otherwise plain b128 load + ds_store.
// ---------------------------------------------------------------------------
#if __has_builtin(__builtin_amdgcn_global_load_async_to_lds_b128) && \
    __has_builtin(__builtin_amdgcn_s_wait_asynccnt)
#define HAVE_ASYNC_LDS 1
#else
#define HAVE_ASYNC_LDS 0
#endif

typedef __attribute__((address_space(1))) v4i* gptr_v4i;   // global
typedef __attribute__((address_space(3))) v4i* lptr_v4i;   // LDS

__device__ __forceinline__ void stage_to_lds(const char* gsrc, char* lds,
                                             int bytes, int tid, int nthr) {
  char* g = const_cast<char*>(gsrc);
  for (int off = tid * 16; off < bytes; off += nthr * 16) {
#if HAVE_ASYNC_LDS
    __builtin_amdgcn_global_load_async_to_lds_b128(
        (gptr_v4i)(g + off), (lptr_v4i)(lds + off), 0, 0);
#else
    *(uint4*)(lds + off) = *(const uint4*)(g + off);
#endif
  }
#if HAVE_ASYNC_LDS
  __builtin_amdgcn_s_wait_asynccnt(0);
#endif
  __syncthreads();
}

// ---------------------------------------------------------------------------
// Generic WMMA GEMM: out[M x N] = A[M x KDIM] * Bcol[N x KDIM]^T (+bias,ReLU)
// 256 threads = 8 wave32; each wave computes a 16x16 tile; blockIdx.y picks
// the 16-row A strip (staged in LDS), blockIdx.x picks 128 columns.
// Output: ND bf16 layout out[(b*TFull + te*TSTEP+TOFF)*ldOut + m], or
// BCT f32 layout out[(b*ldOut + m)*TFull + t] for the final decoder.
// ---------------------------------------------------------------------------
template <int KDIM, bool RELU, bool BCT>
__global__ __launch_bounds__(256) void k_gemm(
    const unsigned short* __restrict__ A,
    const unsigned short* __restrict__ Bc,
    const float* __restrict__ bias,
    unsigned short* __restrict__ outND,
    float* __restrict__ outF32,
    int ldOut, int Tloc, int TSTEP, int TOFF, int TFull) {
  extern __shared__ unsigned short ldsA[];          // 16 * KDIM halves
  const int tid  = threadIdx.x;
  const int lane = tid & 31;
  const int wave = tid >> 5;
  const int hi   = lane >> 4;                       // lane half (wave32)
  const int lm   = lane & 15;
  const int M0   = blockIdx.y * 16;

  stage_to_lds((const char*)(A + (size_t)M0 * KDIM), (char*)ldsA,
               16 * KDIM * 2, tid, 256);

  const int nc = blockIdx.x * 128 + wave * 16 + lm;
  const unsigned short* brow = Bc + (size_t)nc * KDIM;
  const unsigned short* arow = ldsA + lm * KDIM;

  f32x8 acc = {0.f, 0.f, 0.f, 0.f, 0.f, 0.f, 0.f, 0.f};
  for (int k0 = 0; k0 < KDIM; k0 += 32) {
    U32x8 a, b;
    a.lo = *(const uint4*)(arow + k0 + hi * 8);       // K  0..7 / 8..15
    a.hi = *(const uint4*)(arow + k0 + hi * 8 + 16);  // K 16..23 / 24..31
    b.lo = *(const uint4*)(brow + k0 + hi * 8);       // K  0..7 / 8..15
    b.hi = *(const uint4*)(brow + k0 + hi * 8 + 16);  // K 16..23 / 24..31
    acc = __builtin_amdgcn_wmma_f32_16x16x32_bf16(
        false, as_bfrag(a), false, as_bfrag(b), (short)0, acc, false, false);
  }

  const int bb    = nc / Tloc;
  const int te    = nc % Tloc;
  const int t     = te * TSTEP + TOFF;
  const int mbase = M0 + 8 * hi;                    // C/D layout: M = r + 8*hi

  if constexpr (BCT) {
    size_t base = ((size_t)(bb * ldOut + mbase)) * TFull + t;
#pragma unroll
    for (int r = 0; r < 8; ++r) {
      float v = acc[r] + bias[mbase + r];
      if (RELU) v = fmaxf(v, 0.f);
      outF32[base + (size_t)r * TFull] = v;
    }
  } else {
    unsigned short h[8];
#pragma unroll
    for (int r = 0; r < 8; ++r) {
      float v = acc[r] + bias[mbase + r];
      if (RELU) v = fmaxf(v, 0.f);
      h[r] = f32_bf16(v);
    }
    uint4 o = make_uint4((unsigned)h[0] | ((unsigned)h[1] << 16),
                         (unsigned)h[2] | ((unsigned)h[3] << 16),
                         (unsigned)h[4] | ((unsigned)h[5] << 16),
                         (unsigned)h[6] | ((unsigned)h[7] << 16));
    *(uint4*)(outND + ((size_t)(bb * TFull + t)) * ldOut + mbase) = o;
  }
}

// ---------------------------------------------------------------------------
// VQ nearest-code: maximize z.c - 0.5|c|^2 over 512 codes via WMMA.
// Whole bf16 codebook (512x256 = 256KB) staged in LDS (CDNA5: 320KB/WGP).
// ---------------------------------------------------------------------------
__global__ __launch_bounds__(256) void k_vq_argmin(
    const unsigned short* __restrict__ cb,   // [512][256] bf16
    const float* __restrict__ cbias,         // [512] = -0.5*|c|^2
    const unsigned short* __restrict__ z, int ldB,
    int* __restrict__ idxOut) {
  extern __shared__ unsigned short lcb[];
  const int tid = threadIdx.x, lane = tid & 31, wave = tid >> 5;
  const int hi = lane >> 4, lm = lane & 15;

  stage_to_lds((const char*)cb, (char*)lcb, 512 * 256 * 2, tid, 256);

  const int nc = blockIdx.x * 128 + wave * 16 + lm;
  const unsigned short* brow = z + (size_t)nc * ldB;

  float bestv = -3.0e38f;
  int   besti = 0;
  for (int ct = 0; ct < 32; ++ct) {
    const unsigned short* arow = lcb + (ct * 16 + lm) * 256;
    f32x8 acc = {0.f, 0.f, 0.f, 0.f, 0.f, 0.f, 0.f, 0.f};
#pragma unroll
    for (int k0 = 0; k0 < 256; k0 += 32) {
      U32x8 a, b;
      a.lo = *(const uint4*)(arow + k0 + hi * 8);
      a.hi = *(const uint4*)(arow + k0 + hi * 8 + 16);
      b.lo = *(const uint4*)(brow + k0 + hi * 8);
      b.hi = *(const uint4*)(brow + k0 + hi * 8 + 16);
      acc = __builtin_amdgcn_wmma_f32_16x16x32_bf16(
          false, as_bfrag(a), false, as_bfrag(b), (short)0, acc, false, false);
    }
#pragma unroll
    for (int r = 0; r < 8; ++r) {
      int code = ct * 16 + 8 * hi + r;
      float s  = acc[r] + cbias[code];
      if (s > bestv) { bestv = s; besti = code; }
    }
  }
  // lanes L and L^16 hold the same column n -> wave32 xor-shuffle reduce
  float ov = __shfl_xor(bestv, 16, 32);
  int   oi = __shfl_xor(besti, 16, 32);
  if (ov > bestv) { bestv = ov; besti = oi; }
  if (hi == 0) idxOut[nc] = besti;
}

// ---------------------------------------------------------------------------
// im2col builders (all produce K-major bf16 col[N][K])
// ---------------------------------------------------------------------------
// conv1: x (8,64,16384) f32, stride 2, KW=4, pad 1 -> col[65536][256]
__global__ void k_im2col_x(const float* __restrict__ x,
                           unsigned short* __restrict__ col) {
  size_t gid = (size_t)blockIdx.x * 256 + threadIdx.x;
  int k = (int)(gid & 255); size_t n = gid >> 8;
  int b = (int)(n >> 13), t = (int)(n & 8191);
  int i = k >> 2, kw = k & 3, ts = 2 * t + kw - 1;
  float v = 0.f;
  if (ts >= 0 && ts < 16384) v = x[((size_t)(b * 64 + i) << 14) + ts];
  col[gid] = f32_bf16(v);
}

// conv2: h = h1_nd[:,256:512] (ld 512), Tin 8192 -> col[32768][1024]
__global__ void k_im2col_h(const unsigned short* __restrict__ h1,
                           unsigned short* __restrict__ col) {
  size_t gid = (size_t)blockIdx.x * 256 + threadIdx.x;
  int k = (int)(gid & 1023); size_t n = gid >> 10;
  int b = (int)(n >> 12), t = (int)(n & 4095);
  int i = k >> 2, kw = k & 3, ts = 2 * t + kw - 1;
  unsigned short v = 0;
  if (ts >= 0 && ts < 8192) v = h1[((size_t)(b * 8192 + ts)) * 512 + 256 + i];
  col[gid] = v;
}

// convT (stride 2, KW 4, pad 1), parity-split: output t = 2*te + par.
// col[n=(b,te)][i*2+j] = src[b][i][te + par - j], zero OOB.
// src rows are ND with ld 256; channels >=256 come from s2.
__global__ void k_im2colT(const unsigned short* __restrict__ s1,
                          const unsigned short* __restrict__ s2,
                          unsigned short* __restrict__ col,
                          int kshift, int Tin, int par) {
  size_t gid = (size_t)blockIdx.x * 256 + threadIdx.x;
  int K = 1 << kshift;
  int k = (int)(gid & (size_t)(K - 1)); size_t n = gid >> kshift;
  int b = (int)(n / (size_t)Tin), te = (int)(n % (size_t)Tin);
  int i = k >> 1, j = k & 1;
  int tp = te + par - j;
  unsigned short v = 0;
  if (tp >= 0 && tp < Tin) {
    size_t row = ((size_t)(b * Tin + tp)) << 8;   // * 256
    v = (i < 256) ? s1[row + i] : s2[row + (i - 256)];
  }
  col[gid] = v;
}

// ---------------------------------------------------------------------------
// small prep kernels
// ---------------------------------------------------------------------------
__global__ void k_cast_bf16(const float* __restrict__ s,
                            unsigned short* __restrict__ d, int n) {
  int g = blockIdx.x * 256 + threadIdx.x;
  if (g < n) d[g] = f32_bf16(s[g]);
}

__global__ void k_cbnorm(const float* __restrict__ cb, float* __restrict__ o) {
  int c = blockIdx.x * 256 + threadIdx.x;
  if (c < 512) {
    float s = 0.f;
    for (int d = 0; d < 256; ++d) { float v = cb[c * 256 + d]; s += v * v; }
    o[c] = -0.5f * s;
  }
}

// convT weight (I,O,KW) f32 -> parity weight matrix Wp[O][I*2] bf16
// Wp[o][i*2+j] = W[i][o][kwbase + 2*j]
__global__ void k_wT(const float* __restrict__ W,
                     unsigned short* __restrict__ Wp, int I, int O, int kwbase) {
  int g = blockIdx.x * 256 + threadIdx.x;
  if (g >= O * I * 2) return;
  int o = g / (I * 2), r = g % (I * 2), i = r >> 1, j = r & 1;
  Wp[g] = f32_bf16(W[((size_t)i * O + o) * 4 + kwbase + 2 * j]);
}

// zq_nd[n][0..255] = codebook[idx[n]][:]  (straight-through == code row)
__global__ void k_gather(const unsigned short* __restrict__ cb,
                         const int* __restrict__ idx,
                         unsigned short* __restrict__ zq) {
  size_t gid = (size_t)blockIdx.x * 256 + threadIdx.x;
  size_t n = gid >> 5; int c8 = (int)(gid & 31);
  int id = idx[n];
  *(uint4*)(zq + (n << 8) + c8 * 8) =
      *(const uint4*)(cb + ((size_t)id << 8) + c8 * 8);
}

// ---------------------------------------------------------------------------
// launch
// ---------------------------------------------------------------------------
extern "C" void kernel_launch(void* const* d_in, const int* in_sizes, int n_in,
                              void* d_out, int out_size, void* d_ws,
                              size_t ws_size, hipStream_t stream) {
  (void)in_sizes; (void)n_in; (void)out_size; (void)ws_size;
  const float* x      = (const float*)d_in[0];
  const float* W_enc1 = (const float*)d_in[1];
  const float* b_enc1 = (const float*)d_in[2];
  const float* W_enc2 = (const float*)d_in[3];
  const float* b_enc2 = (const float*)d_in[4];
  const float* cb1    = (const float*)d_in[5];
  const float* cb2    = (const float*)d_in[6];
  const float* W_dec1 = (const float*)d_in[7];
  const float* b_dec1 = (const float*)d_in[8];
  const float* W_dec2 = (const float*)d_in[9];
  const float* b_dec2 = (const float*)d_in[10];
  float* out = (float*)d_out;

  char* p = (char*)d_ws;
  auto take = [&](size_t bytes) {
    char* r = p; p += (bytes + 255) & ~(size_t)255; return r;
  };
  unsigned short* h1    = (unsigned short*)take((size_t)65536 * 512 * 2);
  unsigned short* z2    = (unsigned short*)take((size_t)32768 * 256 * 2);
  unsigned short* zq1   = (unsigned short*)take((size_t)65536 * 256 * 2);
  unsigned short* zq2   = (unsigned short*)take((size_t)32768 * 256 * 2);
  unsigned short* d1    = (unsigned short*)take((size_t)65536 * 256 * 2);
  unsigned short* COL   = (unsigned short*)take((size_t)65536 * 1024 * 2);
  unsigned short* Wm1   = (unsigned short*)take(512 * 256 * 2);
  unsigned short* Wm2   = (unsigned short*)take(256 * 1024 * 2);
  unsigned short* Wd1p0 = (unsigned short*)take(256 * 512 * 2);
  unsigned short* Wd1p1 = (unsigned short*)take(256 * 512 * 2);
  unsigned short* Wd2p0 = (unsigned short*)take(64 * 1024 * 2);
  unsigned short* Wd2p1 = (unsigned short*)take(64 * 1024 * 2);
  unsigned short* cb1b  = (unsigned short*)take(512 * 256 * 2);
  unsigned short* cb2b  = (unsigned short*)take(512 * 256 * 2);
  float* cbias1 = (float*)take(512 * 4);
  float* cbias2 = (float*)take(512 * 4);
  int* idx1 = (int*)take(65536 * 4);
  int* idx2 = (int*)take(32768 * 4);

  const dim3 blk(256);

  // ---- weight / codebook prep (bf16 casts, convT parity matrices) ----
  k_cast_bf16<<<512,  blk, 0, stream>>>(W_enc1, Wm1, 131072);
  k_cast_bf16<<<1024, blk, 0, stream>>>(W_enc2, Wm2, 262144);  // first 256 rows
  k_cast_bf16<<<512,  blk, 0, stream>>>(cb1, cb1b, 131072);
  k_cast_bf16<<<512,  blk, 0, stream>>>(cb2, cb2b, 131072);
  k_cbnorm<<<2, blk, 0, stream>>>(cb1, cbias1);
  k_cbnorm<<<2, blk, 0, stream>>>(cb2, cbias2);
  k_wT<<<512, blk, 0, stream>>>(W_dec1, Wd1p0, 256, 256, 1);  // even t: kw 1,3
  k_wT<<<512, blk, 0, stream>>>(W_dec1, Wd1p1, 256, 256, 0);  // odd  t: kw 0,2
  k_wT<<<256, blk, 0, stream>>>(W_dec2, Wd2p0, 512, 64, 1);
  k_wT<<<256, blk, 0, stream>>>(W_dec2, Wd2p1, 512, 64, 0);

  // ---- encoder conv1: 512x256x65536 GEMM, ReLU -> h1_nd[65536][512] ----
  k_im2col_x<<<65536, blk, 0, stream>>>(x, COL);
  k_gemm<256, true, false><<<dim3(512, 32), blk, 16 * 256 * 2, stream>>>(
      Wm1, COL, b_enc1, h1, nullptr, 512, 8192, 1, 0, 8192);

  // ---- encoder conv2 (only z2 channels): 256x1024x32768 GEMM, ReLU ----
  k_im2col_h<<<131072, blk, 0, stream>>>(h1, COL);
  k_gemm<1024, true, false><<<dim3(256, 16), blk, 16 * 1024 * 2, stream>>>(
      Wm2, COL, b_enc2, z2, nullptr, 256, 4096, 1, 0, 4096);

  // ---- vector quantization (WMMA scores + argmax, then gather) ----
  k_vq_argmin<<<512, blk, 512 * 256 * 2, stream>>>(cb1b, cbias1, h1, 512, idx1);
  k_vq_argmin<<<256, blk, 512 * 256 * 2, stream>>>(cb2b, cbias2, z2, 256, idx2);
  k_gather<<<8192, blk, 0, stream>>>(cb1b, idx1, zq1);
  k_gather<<<4096, blk, 0, stream>>>(cb2b, idx2, zq2);

  // ---- decoder convT1 (parity split): 2 x [256x512x32768] GEMM, ReLU ----
  for (int par = 0; par < 2; ++par) {
    k_im2colT<<<65536, blk, 0, stream>>>(zq2, zq2, COL, 9, 4096, par);
    k_gemm<512, true, false><<<dim3(256, 16), blk, 16 * 512 * 2, stream>>>(
        par ? Wd1p1 : Wd1p0, COL, b_dec1, d1, nullptr, 256, 4096, 2, par, 8192);
  }

  // ---- decoder convT2 on concat(d1, zq1): 2 x [64x1024x65536] -> y f32 ----
  for (int par = 0; par < 2; ++par) {
    k_im2colT<<<262144, blk, 0, stream>>>(d1, zq1, COL, 10, 8192, par);
    k_gemm<1024, false, true><<<dim3(512, 4), blk, 16 * 1024 * 2, stream>>>(
        par ? Wd2p1 : Wd2p0, COL, b_dec2, nullptr, out, 64, 8192, 2, par, 16384);
  }
}
